// Attention_71485435675049
// MI455X (gfx1250) — matrix operand order repaired and checked by reading, not compiled, not run on hardware
//
#include <hip/hip_runtime.h>

typedef __attribute__((ext_vector_type(16))) __bf16 v16bf;
typedef __attribute__((ext_vector_type(8)))  __bf16 v8bf;
typedef __attribute__((ext_vector_type(4)))  __bf16 v4bf;
typedef __attribute__((ext_vector_type(2)))  __bf16 v2bf;
typedef __attribute__((ext_vector_type(8)))  float  v8f;
typedef __attribute__((ext_vector_type(4)))  float  v4f;

#define B_    4
#define N_    2048
#define DIM_  1024
#define HEADS 16
#define DHEAD 64
#define INNER 1024
#define NQKV  3072   // 3*INNER

#if defined(__AMDGCN__)
#define HAVE_ASYNC_LDS 1
#else
#define HAVE_ASYNC_LDS 0
#endif

#if HAVE_ASYNC_LDS
// CDNA5 async global->LDS copy (ASYNCcnt-tracked), ISA section 15.18:
//   GLOBAL_LOAD_ASYNC_TO_LDS_B128: VDST = LDS byte address, VADDR = 64b addr.
// Generic pointers to LDS carry the LDS byte offset in their low 32 bits.
__device__ __forceinline__ void async_copy16(const void* g, void* l) {
  unsigned lds_off      = (unsigned)(size_t)l;
  unsigned long long ga = (unsigned long long)(size_t)g;
  asm volatile("global_load_async_to_lds_b128 %0, %1, off"
               :: "v"(lds_off), "v"(ga)
               : "memory");
}
#define ASYNC_WAIT() asm volatile("s_wait_asynccnt 0x0" ::: "memory")
#endif

__device__ __forceinline__ v8f wmma_bf16(v16bf a, v16bf b, v8f c) {
  return __builtin_amdgcn_wmma_f32_16x16x32_bf16(false, a, false, b, (short)0, c,
                                                 false, false);
}

// Concatenate two contiguous 16-byte chunks into one fragment.
__device__ __forceinline__ v16bf frag_cat(const __bf16* p0, const __bf16* p1) {
  v8bf a = *(const v8bf*)p0;
  v8bf b = *(const v8bf*)p1;
  return __builtin_shufflevector(a, b, 0, 1, 2, 3, 4, 5, 6, 7, 8, 9, 10, 11, 12,
                                 13, 14, 15);
}
// A-layout fragment: elements 0..15 = K 2*kb..2*kb+15 (contiguous)
__device__ __forceinline__ v16bf frag_a(const __bf16* row, int kb) {
  return frag_cat(row + 2 * kb, row + 2 * kb + 8);
}
// B-layout fragment from fragment-major tile: 16 contiguous (k = kb2..kb2+15)
__device__ __forceinline__ v16bf frag_b(const __bf16* p) {
  return frag_cat(p, p + 8);
}

// -------------------------------------------------------------------------
// Kernel 1: qkv = x @ w_qkv ; scatter into per-head q/k/v (bf16), q scaled.
// Block tile 64(M) x 128(N); 8 waves; 4 WMMA per wave per K-step (K=32).
// -------------------------------------------------------------------------
__global__ __launch_bounds__(256) void qkv_gemm(const float* __restrict__ x,
                                                const float* __restrict__ w,
                                                __bf16* __restrict__ qb,
                                                __bf16* __restrict__ kb_,
                                                __bf16* __restrict__ vb) {
  __shared__ __bf16 sA[64 * 32];    // [m][k]
  __shared__ __bf16 sBt[128 * 32];  // [n][k]  (fragment-major)
  const int tid  = threadIdx.x;
  const int wid  = tid >> 5;
  const int lane = tid & 31;
  const int m0   = blockIdx.y * 64;
  const int nblk = blockIdx.x * 128;

  const int m    = lane & 15;
  const int half = lane >> 4;
  const int kb   = half * 8;
  const int kb2  = half * 16;
  const int nl   = wid * 16 + m;

  // Per-thread staging coordinates (fixed across K loop)
  const int arow0 = tid >> 3,         ac0 = (tid & 7) * 4;          // A part 0
  const int arow1 = (tid + 256) >> 3, ac1 = ((tid + 256) & 7) * 4;  // A part 1
  const int bkp0 = tid >> 5,          bc0 = (tid & 31) * 4;         // B part 0
  const int bkp1 = (tid + 256) >> 5,  bc1 = ((tid + 256) & 31) * 4; // B part 1

  v8f acc[4] = {};
  for (int kt = 0; kt < DIM_; kt += 32) {
    // ---- issue all global loads first (independent regs -> overlap) ----
    v4f fa0 = *(const v4f*)&x[(size_t)(m0 + arow0) * DIM_ + kt + ac0];
    v4f fa1 = *(const v4f*)&x[(size_t)(m0 + arow1) * DIM_ + kt + ac1];
    v4f fb00 = *(const v4f*)&w[(size_t)(kt + 2 * bkp0) * NQKV + nblk + bc0];
    v4f fb01 = *(const v4f*)&w[(size_t)(kt + 2 * bkp0 + 1) * NQKV + nblk + bc0];
    v4f fb10 = *(const v4f*)&w[(size_t)(kt + 2 * bkp1) * NQKV + nblk + bc1];
    v4f fb11 = *(const v4f*)&w[(size_t)(kt + 2 * bkp1 + 1) * NQKV + nblk + bc1];
    if (kt + 32 < DIM_)
      __builtin_prefetch(&w[(size_t)(kt + 32 + (tid >> 5)) * NQKV + nblk + (tid & 31) * 4], 0, 0);

    // ---- convert + store ----
    {
      v4bf bc;
      bc[0] = (__bf16)fa0[0]; bc[1] = (__bf16)fa0[1];
      bc[2] = (__bf16)fa0[2]; bc[3] = (__bf16)fa0[3];
      *(v4bf*)&sA[arow0 * 32 + ac0] = bc;
      bc[0] = (__bf16)fa1[0]; bc[1] = (__bf16)fa1[1];
      bc[2] = (__bf16)fa1[2]; bc[3] = (__bf16)fa1[3];
      *(v4bf*)&sA[arow1 * 32 + ac1] = bc;
    }
#pragma unroll
    for (int j = 0; j < 4; ++j) {
      v2bf p;
      p[0] = (__bf16)fb00[j]; p[1] = (__bf16)fb01[j];
      *(v2bf*)&sBt[(bc0 + j) * 32 + 2 * bkp0] = p;
      p[0] = (__bf16)fb10[j]; p[1] = (__bf16)fb11[j];
      *(v2bf*)&sBt[(bc1 + j) * 32 + 2 * bkp1] = p;
    }
    __syncthreads();

    v16bf bfrag = frag_b(&sBt[nl * 32 + kb2]);
#pragma unroll
    for (int mt = 0; mt < 4; ++mt) {
      v16bf afrag = frag_a(&sA[(mt * 16 + m) * 32], kb);
      acc[mt] = wmma_bf16(afrag, bfrag, acc[mt]);
    }
    __syncthreads();
  }

  // Epilogue: scatter into q/k/v [bh, n, d]
  const int nc    = nblk + nl;
  const int which = nc / INNER;
  const int inner = nc % INNER;
  const int h     = inner / DHEAD;
  const int d     = inner % DHEAD;
  __bf16* dst     = (which == 0) ? qb : ((which == 1) ? kb_ : vb);
  const float scl = (which == 0) ? 0.125f : 1.0f;  // DIM_HEAD^-0.5
#pragma unroll
  for (int mt = 0; mt < 4; ++mt) {
#pragma unroll
    for (int r = 0; r < 8; ++r) {
      int g  = m0 + mt * 16 + r + 8 * half;  // global row in [0, B*N)
      int b  = g / N_;
      int np = g % N_;
      dst[((size_t)(b * HEADS + h) * N_ + np) * DHEAD + d] =
          (__bf16)(acc[mt][r] * scl);
    }
  }
}

// -------------------------------------------------------------------------
// Kernel 2: flash attention. grid = (64 bh, 16 qchunks); 8 waves x 16 rows.
// -------------------------------------------------------------------------
__global__ __launch_bounds__(256) void attn_kernel(const __bf16* __restrict__ q,
                                                   const __bf16* __restrict__ k,
                                                   const __bf16* __restrict__ v,
                                                   __bf16* __restrict__ ao) {
  __shared__ __bf16 sK[32 * 64];    // [j][d] (fragment-major for sim)
  __shared__ __bf16 sVt[64 * 32];   // [d][j] (fragment-major for PV)
  __shared__ __bf16 sP[8][16 * 32];

  const int tid  = threadIdx.x;
  const int wid  = tid >> 5;
  const int lane = tid & 31;
  const int bh   = blockIdx.x;
  const int q0   = blockIdx.y * 128 + wid * 16;
  const size_t base = (size_t)bh * N_ * DHEAD;

  const int m    = lane & 15;
  const int half = lane >> 4;
  const int kb   = half * 8;
  const int kb2  = half * 16;

  // q A-fragments (K=64 -> two 16x32 fragments), contiguous 16B global reads.
  v16bf qf[2];
#pragma unroll
  for (int w2 = 0; w2 < 2; ++w2)
    qf[w2] = frag_a(q + base + (size_t)(q0 + m) * DHEAD + w2 * 32, kb);

  float mrow[8], lrow[8];
#pragma unroll
  for (int r = 0; r < 8; ++r) { mrow[r] = -__builtin_inff(); lrow[r] = 0.0f; }
  v8f o[4] = {};

  // V-transpose staging coords
  const int vjp = tid >> 4;        // row pair 0..15
  const int vd0 = (tid & 15) * 4;  // d group of 4

  for (int j0 = 0; j0 < N_; j0 += 32) {
    // K chunk: straight 16B/lane copy -> CDNA5 async LDS load
#if HAVE_ASYNC_LDS
    async_copy16(k + base + (size_t)j0 * DHEAD + tid * 8, &sK[tid * 8]);
#else
    ((uint4*)sK)[tid] = ((const uint4*)(k + base + (size_t)j0 * DHEAD))[tid];
#endif
    // V chunk: transpose to sVt[d][j]; pack (j, j+1) pairs -> 32-bit stores
    {
      const __bf16* vr0 = v + base + (size_t)(j0 + 2 * vjp) * DHEAD + vd0;
      v4bf t0 = *(const v4bf*)vr0;
      v4bf t1 = *(const v4bf*)(vr0 + DHEAD);
#pragma unroll
      for (int i = 0; i < 4; ++i) {
        v2bf p;
        p[0] = t0[i]; p[1] = t1[i];
        *(v2bf*)&sVt[(vd0 + i) * 32 + 2 * vjp] = p;
      }
    }
#if HAVE_ASYNC_LDS
    ASYNC_WAIT();
#endif
    __syncthreads();

    // sim = q @ K^T for 32 keys -> two 16x16 tiles
    v8f sim[2];
#pragma unroll
    for (int jt = 0; jt < 2; ++jt) {
      const __bf16* kp = &sK[(jt * 16 + m) * 64];
      v8f c = {};
      c = wmma_bf16(qf[0], frag_b(kp + kb2), c);
      c = wmma_bf16(qf[1], frag_b(kp + 32 + kb2), c);
      sim[jt] = c;
    }

    // Online softmax: C layout -> row = r + 8*half, col = lane&15.
#pragma unroll
    for (int r = 0; r < 8; ++r) {
      float t = fmaxf(sim[0][r], sim[1][r]);
#pragma unroll
      for (int msk = 8; msk >= 1; msk >>= 1)
        t = fmaxf(t, __shfl_xor(t, msk, 32));
      float mn    = fmaxf(mrow[r], t);
      float alpha = __expf(mrow[r] - mn);
      float p0    = __expf(sim[0][r] - mn);
      float p1    = __expf(sim[1][r] - mn);
      float s     = p0 + p1;
#pragma unroll
      for (int msk = 8; msk >= 1; msk >>= 1)
        s += __shfl_xor(s, msk, 32);
      lrow[r] = lrow[r] * alpha + s;
      mrow[r] = mn;
#pragma unroll
      for (int t4 = 0; t4 < 4; ++t4) o[t4][r] *= alpha;
      const int prow = r + 8 * half;
      sP[wid][prow * 32 + m]      = (__bf16)p0;
      sP[wid][prow * 32 + 16 + m] = (__bf16)p1;
    }

    // P (16x32) A-fragment from wave-private LDS (cross-lane re-layout).
    v16bf pf = frag_a(&sP[wid][m * 32], kb);
    // o += P @ V : four 16x16 d-tiles, B-fragments contiguous from sVt
#pragma unroll
    for (int t4 = 0; t4 < 4; ++t4) {
      v16bf vf = frag_b(&sVt[(t4 * 16 + m) * 32 + kb2]);
      o[t4] = wmma_bf16(pf, vf, o[t4]);
    }
    __syncthreads();
  }

  // Epilogue: normalize and write bf16 to [b, n, h*64+d]
  const int b = bh / HEADS, h = bh % HEADS;
#pragma unroll
  for (int t4 = 0; t4 < 4; ++t4) {
#pragma unroll
    for (int r = 0; r < 8; ++r) {
      int np    = q0 + r + 8 * half;
      float val = o[t4][r] / lrow[r];
      ao[((size_t)b * N_ + np) * INNER + h * DHEAD + t4 * 16 + m] = (__bf16)val;
    }
  }
}

// -------------------------------------------------------------------------
// Kernel 3: out = attn @ w_out + b_out (fp32 out).
// -------------------------------------------------------------------------
__global__ __launch_bounds__(256) void out_gemm(const __bf16* __restrict__ a,
                                                const float* __restrict__ w,
                                                const float* __restrict__ bias,
                                                float* __restrict__ out) {
  __shared__ __bf16 sA[64 * 32];    // [m][k]
  __shared__ __bf16 sBt[128 * 32];  // [n][k]
  const int tid  = threadIdx.x;
  const int wid  = tid >> 5;
  const int lane = tid & 31;
  const int m0   = blockIdx.y * 64;
  const int nblk = blockIdx.x * 128;

  const int m    = lane & 15;
  const int half = lane >> 4;
  const int kb   = half * 8;
  const int kb2  = half * 16;
  const int nl   = wid * 16 + m;

  const int arow = tid >> 2, ac8 = (tid & 3) * 8;
  const int bkp0 = tid >> 5,         bc0 = (tid & 31) * 4;
  const int bkp1 = (tid + 256) >> 5, bc1 = ((tid + 256) & 31) * 4;

  v8f acc[4] = {};
  for (int kt = 0; kt < INNER; kt += 32) {
    // A tile 64x32 bf16: straight 16B/lane copy -> async LDS load
#if HAVE_ASYNC_LDS
    async_copy16(&a[(size_t)(m0 + arow) * INNER + kt + ac8], &sA[arow * 32 + ac8]);
#else
    *(uint4*)&sA[arow * 32 + ac8] =
        *(const uint4*)&a[(size_t)(m0 + arow) * INNER + kt + ac8];
#endif
    // B tile 32x128 f32 -> transposed [n][k], packed (k,k+1) 32-bit stores
    v4f fb00 = *(const v4f*)&w[(size_t)(kt + 2 * bkp0) * DIM_ + nblk + bc0];
    v4f fb01 = *(const v4f*)&w[(size_t)(kt + 2 * bkp0 + 1) * DIM_ + nblk + bc0];
    v4f fb10 = *(const v4f*)&w[(size_t)(kt + 2 * bkp1) * DIM_ + nblk + bc1];
    v4f fb11 = *(const v4f*)&w[(size_t)(kt + 2 * bkp1 + 1) * DIM_ + nblk + bc1];
    if (kt + 32 < INNER)
      __builtin_prefetch(&w[(size_t)(kt + 32 + (tid >> 5)) * DIM_ + nblk + (tid & 31) * 4], 0, 0);
#pragma unroll
    for (int j = 0; j < 4; ++j) {
      v2bf p;
      p[0] = (__bf16)fb00[j]; p[1] = (__bf16)fb01[j];
      *(v2bf*)&sBt[(bc0 + j) * 32 + 2 * bkp0] = p;
      p[0] = (__bf16)fb10[j]; p[1] = (__bf16)fb11[j];
      *(v2bf*)&sBt[(bc1 + j) * 32 + 2 * bkp1] = p;
    }
#if HAVE_ASYNC_LDS
    ASYNC_WAIT();
#endif
    __syncthreads();

    v16bf bfrag = frag_b(&sBt[nl * 32 + kb2]);
#pragma unroll
    for (int mt = 0; mt < 4; ++mt) {
      v16bf afrag = frag_a(&sA[(mt * 16 + m) * 32], kb);
      acc[mt] = wmma_bf16(afrag, bfrag, acc[mt]);
    }
    __syncthreads();
  }

  const int nc   = nblk + nl;
  const float bv = bias[nc];
#pragma unroll
  for (int mt = 0; mt < 4; ++mt) {
#pragma unroll
    for (int r = 0; r < 8; ++r) {
      int g = m0 + mt * 16 + r + 8 * half;
      out[(size_t)g * DIM_ + nc] = acc[mt][r] + bv;
    }
  }
}

// -------------------------------------------------------------------------
extern "C" void kernel_launch(void* const* d_in, const int* in_sizes, int n_in,
                              void* d_out, int out_size, void* d_ws, size_t ws_size,
                              hipStream_t stream) {
  const float* x     = (const float*)d_in[0];
  const float* w_qkv = (const float*)d_in[1];
  const float* w_out = (const float*)d_in[2];
  const float* b_out = (const float*)d_in[3];
  float* out         = (float*)d_out;

  char* ws = (char*)d_ws;
  const size_t per = (size_t)B_ * HEADS * N_ * DHEAD * sizeof(__bf16);  // 16 MB
  __bf16* qb = (__bf16*)(ws);
  __bf16* kb = (__bf16*)(ws + per);
  __bf16* vb = (__bf16*)(ws + 2 * per);
  __bf16* ab = (__bf16*)(ws + 3 * per);

  // 1) qkv projection: [8192x1024] @ [1024x3072]
  qkv_gemm<<<dim3(NQKV / 128, (B_ * N_) / 64), 256, 0, stream>>>(x, w_qkv, qb, kb, vb);
  // 2) flash attention over 64 (b,h) batches
  attn_kernel<<<dim3(B_ * HEADS, N_ / 128), 256, 0, stream>>>(qb, kb, vb, ab);
  // 3) output projection + bias: [8192x1024] @ [1024x1024]
  out_gemm<<<dim3(DIM_ / 128, (B_ * N_) / 64), 256, 0, stream>>>(ab, w_out, b_out, out);
}
